// SlidingWindowAttention_28965259444851
// MI455X (gfx1250) — matrix operand order repaired and checked by reading, not compiled, not run on hardware
//
#include <hip/hip_runtime.h>
#include <hip/hip_bf16.h>

// Problem constants (match reference)
#define BB   2
#define TT   2048
#define BDIM 1024
#define NH   16
#define HD   64
#define WIN  256

typedef __attribute__((ext_vector_type(16))) __bf16 v16bf;
typedef __attribute__((ext_vector_type(8)))  float  v8f;
typedef __attribute__((ext_vector_type(4)))  int    v4i;

__device__ inline v8f v8f_zero() {
    v8f v;
#pragma unroll
    for (int i = 0; i < 8; ++i) v[i] = 0.0f;
    return v;
}

// ---------------------------------------------------------------------------
// Async global->LDS staging (gfx1250 GLOBAL_LOAD_ASYNC_TO_LDS_B128, ASYNCcnt)
// with a synchronous uint4 fallback so both compile passes stay clean.
// Builtin param 1 is `v4i __device__(AS1)*` per the round-2 diagnostic.
// ---------------------------------------------------------------------------
#if defined(__AMDGCN__) &&                                            \
    __has_builtin(__builtin_amdgcn_global_load_async_to_lds_b128) &&  \
    __has_builtin(__builtin_amdgcn_s_wait_asynccnt)
#define ASYNC_LDS 1
#else
#define ASYNC_LDS 0
#endif

__device__ inline void stage16(__bf16* lds_dst, const __bf16* gsrc) {
#if ASYNC_LDS
    __bf16* g = const_cast<__bf16*>(gsrc);
    __builtin_amdgcn_global_load_async_to_lds_b128(
        (__attribute__((address_space(1))) v4i*)g,
        (__attribute__((address_space(3))) v4i*)lds_dst, 0, 0);
#else
    *reinterpret_cast<uint4*>(lds_dst) = *reinterpret_cast<const uint4*>(gsrc);
#endif
}

__device__ inline void stage_wait() {
#if ASYNC_LDS
    __builtin_amdgcn_s_wait_asynccnt(0);
#endif
}

// ---------------------------------------------------------------------------
// f32 -> bf16 cast
// ---------------------------------------------------------------------------
__global__ __launch_bounds__(256) void cast_bf16(const float* __restrict__ in,
                                                 __bf16* __restrict__ out, int n) {
    int i = blockIdx.x * blockDim.x + threadIdx.x;
    if (i < n) out[i] = (__bf16)in[i];
}

// ---------------------------------------------------------------------------
// Tiled WMMA GEMM:  Out(M,N) = A(M,K) @ W(N,K)^T,  bf16 in, f32 accumulate.
// Block: 256 threads = 8 waves; block tile 128x256, K-step 32, double-buffered
// async LDS staging. Wave grid 2(M) x 4(N); each wave computes 64x64 = 4x4
// WMMA tiles (16 v_wmma per k-step per wave).
// ---------------------------------------------------------------------------
template <bool OUT_BF16>
__global__ __launch_bounds__(256) void wmma_gemm(const __bf16* __restrict__ A,
                                                 const __bf16* __restrict__ W,
                                                 float* __restrict__ outF,
                                                 __bf16* __restrict__ outB,
                                                 int M, int N, int K) {
    __shared__ __align__(16) __bf16 sA[2][128][40];  // 80B row stride (16B-aligned chunks)
    __shared__ __align__(16) __bf16 sB[2][256][32];  // row = n, col = k

    const int tid   = threadIdx.x;
    const int lane  = tid & 31;
    const int wave  = tid >> 5;
    const int m0    = blockIdx.x * 128;
    const int n0    = blockIdx.y * 256;
    const int waveM = wave >> 2;      // 0..1
    const int waveN = wave & 3;       // 0..3
    const int laneM = lane & 15;
    const int hi    = lane >> 4;      // 0 or 1
    const int kbase = hi * 8;         // A-frag K base per ISA layout

    v8f acc[4][4];
#pragma unroll
    for (int i = 0; i < 4; ++i)
#pragma unroll
        for (int j = 0; j < 4; ++j) acc[i][j] = v8f_zero();

    const int ldr = tid >> 1;         // 0..127 (tile row to stage)
    const int ldc = (tid & 1) * 16;   // 0 or 16 (tile col base)

    // Stage one K-tile (A: 128x32, B: 256x32) into LDS buffer `buf`.
    auto stageA = [&](int buf, int k0) {
        stage16(&sA[buf][ldr][ldc], A + (size_t)(m0 + ldr) * K + k0 + ldc);
    };
    auto stageB = [&](int buf, int k0) {
        stage16(&sB[buf][ldr][ldc],
                W + (size_t)(n0 + ldr) * K + k0 + ldc);
        stage16(&sB[buf][ldr + 128][ldc],
                W + (size_t)(n0 + 128 + ldr) * K + k0 + ldc);
    };

    stageA(0, 0);
    stageB(0, 0);

    int buf = 0;
    for (int k0 = 0; k0 < K; k0 += 32) {
        stage_wait();       // staged tile for this iteration has landed (per wave)
        __syncthreads();    // ...and is visible block-wide
        if (k0 + 32 < K) {  // overlap: stream next tile while computing
            stageA(buf ^ 1, k0 + 32);
            stageB(buf ^ 1, k0 + 32);
        }

        // --- A fragments (16x32, ISA layout: lane<16 K in {0..7,16..23}) ---
        v16bf afrag[4];
#pragma unroll
        for (int mi = 0; mi < 4; ++mi) {
            union { uint4 u[2]; v16bf v; } t;
            const __bf16* base = &sA[buf][waveM * 64 + mi * 16 + laneM][kbase];
            t.u[0] = *reinterpret_cast<const uint4*>(base);
            t.u[1] = *reinterpret_cast<const uint4*>(base + 16);
            afrag[mi] = t.v;
        }
        // --- B fragments (32x16, N = laneM, element = K in A's table order) ---
        v16bf bfrag[4];
#pragma unroll
        for (int ni = 0; ni < 4; ++ni) {
            v16bf b;
#pragma unroll
            for (int e = 0; e < 16; ++e)
                b[e] = sB[buf][waveN * 64 + ni * 16 + laneM][kbase + (e & 7) + ((e >> 3) << 4)];
            bfrag[ni] = b;
        }
#pragma unroll
        for (int mi = 0; mi < 4; ++mi)
#pragma unroll
            for (int ni = 0; ni < 4; ++ni)
                acc[mi][ni] = __builtin_amdgcn_wmma_f32_16x16x32_bf16(
                    false, afrag[mi], false, bfrag[ni], (short)0, acc[mi][ni],
                    false, false);

        __syncthreads();    // all waves done reading `buf` before it is re-staged
        buf ^= 1;
    }

    // --- epilogue: C layout row = r + hi*8, col = laneM ---
#pragma unroll
    for (int mi = 0; mi < 4; ++mi)
#pragma unroll
        for (int ni = 0; ni < 4; ++ni)
#pragma unroll
            for (int r = 0; r < 8; ++r) {
                int row = m0 + waveM * 64 + mi * 16 + r + hi * 8;
                int col = n0 + waveN * 64 + ni * 16 + laneM;
                float v = acc[mi][ni][r];
                if (OUT_BF16)
                    outB[(size_t)row * N + col] = (__bf16)v;
                else
                    outF[(size_t)row * N + col] = v;
            }
}

// ---------------------------------------------------------------------------
// Sliding-window causal flash attention.
// One wave handles one (b, h, 16-query tile). qkv layout (B, T, 3*DIM) bf16.
// ---------------------------------------------------------------------------
__global__ __launch_bounds__(256) void swa_attn(const __bf16* __restrict__ qkv,
                                                __bf16* __restrict__ out) {
    __shared__ __align__(16) __bf16 sP[8][16][40];  // per-wave P bounce buffer

    const int tid  = threadIdx.x;
    const int lane = tid & 31;
    const int wave = tid >> 5;
    const int gw   = blockIdx.x * 8 + wave;
    const int T16  = TT / 16;
    const int qb   = gw % T16;
    const int h    = (gw / T16) % NH;
    const int b    = gw / (T16 * NH);
    const int q0   = qb * 16;

    const int laneM = lane & 15;
    const int hi    = lane >> 4;
    const int kbase = hi * 8;
    const size_t rs = 3 * BDIM;  // qkv row stride (elements)

    // --- Q fragments: A layout, row = laneM, two K-chunks of 32 over hd=64 ---
    const __bf16* qbase = qkv + ((size_t)b * TT + q0) * rs + (size_t)h * HD;
    v16bf qf[2];
#pragma unroll
    for (int kk = 0; kk < 2; ++kk) {
        union { uint4 u[2]; v16bf v; } t;
        const __bf16* p = qbase + (size_t)laneM * rs + kk * 32 + kbase;
        t.u[0] = *reinterpret_cast<const uint4*>(p);       // K = kbase..kbase+7
        t.u[1] = *reinterpret_cast<const uint4*>(p + 16);  // K = kbase+16..kbase+23
        qf[kk] = t.v;
    }

    float mrow[8], lrow[8];
    v8f   oacc[4];
#pragma unroll
    for (int r = 0; r < 8; ++r) { mrow[r] = -__builtin_inff(); lrow[r] = 0.0f; }
#pragma unroll
    for (int t = 0; t < 4; ++t) oacc[t] = v8f_zero();

    int js = q0 - (WIN - 1);
    if (js < 0) js = 0;
    js &= ~31;

    for (int j0 = js; j0 <= q0 + 15; j0 += 32) {
        // ---- scores: two 16-key groups, K-dim = 64 (2 WMMAs each) ----
        v8f s[2];
#pragma unroll
        for (int g = 0; g < 2; ++g) {
            v8f sv = v8f_zero();
#pragma unroll
            for (int kk = 0; kk < 2; ++kk) {
                // B fragment: key = j0+g*16+laneM, element e = hd index (table order)
                v16bf kf;
                const __bf16* kb = qkv + ((size_t)b * TT + j0 + g * 16 + laneM) * rs +
                                   BDIM + (size_t)h * HD + kk * 32 + kbase;
#pragma unroll
                for (int e = 0; e < 16; ++e) kf[e] = kb[(e & 7) + ((e >> 3) << 4)];
                sv = __builtin_amdgcn_wmma_f32_16x16x32_bf16(
                    false, qf[kk], false, kf, (short)0, sv, false, false);
            }
            s[g] = sv;
        }
        // ---- scale + sliding-window causal mask ----
#pragma unroll
        for (int g = 0; g < 2; ++g)
#pragma unroll
            for (int r = 0; r < 8; ++r) {
                int i = q0 + r + hi * 8;
                int j = j0 + g * 16 + laneM;
                float v = s[g][r] * 0.125f;  // 1/sqrt(64)
                bool masked = (j > i) || (j <= i - WIN);
                s[g][r] = masked ? -__builtin_inff() : v;
            }
        // ---- online softmax (row spread across 16-lane half) ----
        float corr[8];
#pragma unroll
        for (int r = 0; r < 8; ++r) {
            float mx = fmaxf(s[0][r], s[1][r]);
#pragma unroll
            for (int off = 8; off >= 1; off >>= 1)
                mx = fmaxf(mx, __shfl_xor(mx, off, 32));
            float mnew = fmaxf(mrow[r], mx);
            corr[r] = (mnew == -__builtin_inff()) ? 1.0f : expf(mrow[r] - mnew);
            mrow[r] = mnew;
            float p0 = (s[0][r] == -__builtin_inff()) ? 0.0f : expf(s[0][r] - mnew);
            float p1 = (s[1][r] == -__builtin_inff()) ? 0.0f : expf(s[1][r] - mnew);
            s[0][r] = p0;
            s[1][r] = p1;
            float ps = p0 + p1;
#pragma unroll
            for (int off = 8; off >= 1; off >>= 1)
                ps += __shfl_xor(ps, off, 32);
            lrow[r] = lrow[r] * corr[r] + ps;
        }
#pragma unroll
        for (int t = 0; t < 4; ++t)
#pragma unroll
            for (int r = 0; r < 8; ++r) oacc[t][r] *= corr[r];

        // ---- P: C-layout -> A-layout via per-wave LDS bounce (LDS in-order) ----
#pragma unroll
        for (int g = 0; g < 2; ++g)
#pragma unroll
            for (int r = 0; r < 8; ++r)
                sP[wave][r + hi * 8][g * 16 + laneM] = (__bf16)s[g][r];
        v16bf pf;
        {
            union { uint4 u[2]; v16bf v; } t;
            const __bf16* base = &sP[wave][laneM][kbase];
            t.u[0] = *reinterpret_cast<const uint4*>(base);
            t.u[1] = *reinterpret_cast<const uint4*>(base + 16);
            pf = t.v;
        }
        // ---- O += P(16x32) @ V(32x64): 4 WMMAs ----
#pragma unroll
        for (int t4 = 0; t4 < 4; ++t4) {
            // B fragment: hd col = t4*16+laneM, element e = key (table order)
            union { uint4 u[2]; v16bf v; } tv;
            const __bf16* vb = qkv + ((size_t)b * TT + j0 + kbase) * rs + 2 * BDIM +
                               (size_t)h * HD + t4 * 16 + laneM;
#pragma unroll
            for (int e = 0; e < 16; ++e)
                tv.v[e] = vb[(size_t)((e & 7) + ((e >> 3) << 4)) * rs];
            oacc[t4] = __builtin_amdgcn_wmma_f32_16x16x32_bf16(
                false, pf, false, tv.v, (short)0, oacc[t4], false, false);
        }
    }

    // ---- epilogue: out[b, q0+row, h*64 + d] = O / l ----
    __bf16* ob = out + ((size_t)b * TT + q0) * BDIM + (size_t)h * HD;
#pragma unroll
    for (int r = 0; r < 8; ++r) {
        float inv = (lrow[r] > 0.0f) ? (1.0f / lrow[r]) : 0.0f;
#pragma unroll
        for (int t4 = 0; t4 < 4; ++t4)
            ob[(size_t)(r + hi * 8) * BDIM + t4 * 16 + laneM] =
                (__bf16)(oacc[t4][r] * inv);
    }
}

// ---------------------------------------------------------------------------
// Launcher
// ---------------------------------------------------------------------------
extern "C" void kernel_launch(void* const* d_in, const int* in_sizes, int n_in,
                              void* d_out, int out_size, void* d_ws, size_t ws_size,
                              hipStream_t stream) {
    (void)in_sizes; (void)n_in; (void)out_size; (void)ws_size;
    const float* x    = (const float*)d_in[0];
    const float* Wqkv = (const float*)d_in[1];
    const float* Wout = (const float*)d_in[2];
    float* out = (float*)d_out;

    char* ws = (char*)d_ws;
    __bf16* xb  = (__bf16*)(ws);                          //  8 MiB: x bf16
    __bf16* wqb = (__bf16*)(ws + ((size_t)8  << 20));     //  6 MiB: W_qkv bf16
    __bf16* wob = (__bf16*)(ws + ((size_t)14 << 20));     //  2 MiB: W_out bf16
    __bf16* qkv = (__bf16*)(ws + ((size_t)16 << 20));     // 24 MiB: qkv bf16
    __bf16* att = (__bf16*)(ws + ((size_t)40 << 20));     //  8 MiB: attn out bf16

    const int nx = BB * TT * BDIM;      // 4194304
    const int nq = 3 * BDIM * BDIM;     // 3145728
    const int no = BDIM * BDIM;         // 1048576
    cast_bf16<<<(nx + 255) / 256, 256, 0, stream>>>(x, xb, nx);
    cast_bf16<<<(nq + 255) / 256, 256, 0, stream>>>(Wqkv, wqb, nq);
    cast_bf16<<<(no + 255) / 256, 256, 0, stream>>>(Wout, wob, no);

    // qkv = x @ W_qkv^T  (4096 x 3072)
    dim3 g1(BB * TT / 128, 3 * BDIM / 256);
    wmma_gemm<true><<<g1, 256, 0, stream>>>(xb, wqb, nullptr, qkv,
                                            BB * TT, 3 * BDIM, BDIM);

    // sliding-window attention (4096 query tiles, 8 waves/block)
    swa_attn<<<(BB * NH * (TT / 16)) / 8, 256, 0, stream>>>(qkv, att);

    // out = att @ W_out^T  (4096 x 1024), f32 output
    dim3 g2(BB * TT / 128, BDIM / 256);
    wmma_gemm<false><<<g2, 256, 0, stream>>>(att, wob, out, nullptr,
                                             BB * TT, BDIM, BDIM);
}
